// ASRModel_Attention_20186346292017
// MI455X (gfx1250) — compile-verified, matrix operand order checked by reading, not run on hardware
//
#include <hip/hip_runtime.h>
#include <stdint.h>

// Problem constants (from reference setup_inputs)
#define B_  64
#define S_  1024
#define E_  512
#define H_  512
#define HK_ 256
#define HV_ 512
#define T_  300
#define V_  5000
#define G4_ 2048   // 4*H gate width

// ---------------- CDNA5 WMMA plumbing ----------------
typedef __attribute__((ext_vector_type(16))) __bf16       v16bf;
typedef __attribute__((ext_vector_type(8)))  float        v8f;
typedef __attribute__((ext_vector_type(4)))  unsigned int u32x4;

union FragU { v16bf v; u32x4 q[2]; };

__device__ __forceinline__ unsigned short f2bf(float f) {
  unsigned int u = __float_as_uint(f);
  u += 0x7FFFu + ((u >> 16) & 1u);           // round-to-nearest-even
  return (unsigned short)(u >> 16);
}
__device__ __forceinline__ float bf2f(unsigned short u) {
  return __uint_as_float(((unsigned int)u) << 16);
}
__device__ __forceinline__ float sigmoidf_(float x) {
  return 1.0f / (1.0f + __expf(-x));
}

// Fragment gather at a per-lane pointer: two 16B chunks 32B apart
// (K = kbase+half*8 .. +7 and K = kbase+16+half*8 .. +7, per ISA 16-bit A/B
// layout in CDNA5 ISA). Pointer advances 4 u32x4 (= 32 bf16 of K) per step.
__device__ __forceinline__ v16bf frag_at(const u32x4* __restrict__ p) {
  FragU f;
  f.q[0] = p[0];
  f.q[1] = p[2];
  return f.v;
}

__device__ __forceinline__ v8f wmma_bf16(v16bf a, v16bf b, v8f c) {
  // (neg_a, A, neg_b, B, c_mod, C, reuse_a, reuse_b)
  return __builtin_amdgcn_wmma_f32_16x16x32_bf16(false, a, false, b,
                                                 (short)0, c, false, false);
}

// Software-pipelined K-strip: 1 A-tile x 4 N-tiles. Next iteration's 10
// global_load_b128 are issued before the current 4 WMMAs consume their
// fragments, so loads overlap the matrix pipe instead of draining loadcnt
// before every v_wmma.
__device__ __forceinline__ void gemm_strip(const u32x4* __restrict__ pa,
                                           const u32x4* __restrict__ pb0,
                                           const u32x4* __restrict__ pb1,
                                           const u32x4* __restrict__ pb2,
                                           const u32x4* __restrict__ pb3,
                                           int K,
                                           v8f& acc0, v8f& acc1,
                                           v8f& acc2, v8f& acc3) {
  v16bf a  = frag_at(pa);
  v16bf b0 = frag_at(pb0);
  v16bf b1 = frag_at(pb1);
  v16bf b2 = frag_at(pb2);
  v16bf b3 = frag_at(pb3);
  for (int k = 32; k < K; k += 32) {
    pa += 4; pb0 += 4; pb1 += 4; pb2 += 4; pb3 += 4;
    const v16bf an  = frag_at(pa);     // prefetch next K-slice (independent)
    const v16bf bn0 = frag_at(pb0);
    const v16bf bn1 = frag_at(pb1);
    const v16bf bn2 = frag_at(pb2);
    const v16bf bn3 = frag_at(pb3);
    acc0 = wmma_bf16(a, b0, acc0);     // consume current fragments
    acc1 = wmma_bf16(a, b1, acc1);
    acc2 = wmma_bf16(a, b2, acc2);
    acc3 = wmma_bf16(a, b3, acc3);
    a = an; b0 = bn0; b1 = bn1; b2 = bn2; b3 = bn3;
  }
  acc0 = wmma_bf16(a, b0, acc0);       // epilogue
  acc1 = wmma_bf16(a, b1, acc1);
  acc2 = wmma_bf16(a, b2, acc2);
  acc3 = wmma_bf16(a, b3, acc3);
}

// ---------------- Kernels ----------------

__global__ void cast_f32_bf16_kernel(const float* __restrict__ in,
                                     unsigned short* __restrict__ outp,
                                     size_t n) {
  size_t i = (size_t)blockIdx.x * blockDim.x + threadIdx.x;
  size_t stride = (size_t)gridDim.x * blockDim.x;
  for (; i < n; i += stride) outp[i] = f2bf(in[i]);
}

// Generic WMMA GEMM, bf16 out: Out[M,N] = act(A[M,K] @ W[N,K]^T + bias[N]).
// block = 128 (4 waves). Wave = 1 M-tile x 4 N-tiles (A-fragment reuse).
// grid = (N/64, M/64). M, N multiples of 64; no load guards needed.
__global__ void gemm_bf16out_kernel(const unsigned short* __restrict__ A,
                                    const unsigned short* __restrict__ W,
                                    const float* __restrict__ bias,
                                    unsigned short* __restrict__ Out,
                                    int M, int N, int K, int act) {
  const int lane = threadIdx.x & 31;
  const int half = lane >> 4, idx = lane & 15;
  const int mt   = blockIdx.y * 4 + (threadIdx.x >> 5);
  const int nb   = blockIdx.x * 64;
  const int ko   = half * 8;

  const u32x4* pa  = (const u32x4*)(A + (size_t)(mt * 16 + idx) * K + ko);
  const u32x4* pb0 = (const u32x4*)(W + (size_t)(nb + 0  + idx) * K + ko);
  const u32x4* pb1 = (const u32x4*)(W + (size_t)(nb + 16 + idx) * K + ko);
  const u32x4* pb2 = (const u32x4*)(W + (size_t)(nb + 32 + idx) * K + ko);
  const u32x4* pb3 = (const u32x4*)(W + (size_t)(nb + 48 + idx) * K + ko);

  v8f acc0 = {}, acc1 = {}, acc2 = {}, acc3 = {};
  gemm_strip(pa, pb0, pb1, pb2, pb3, K, acc0, acc1, acc2, acc3);

#pragma unroll
  for (int j = 0; j < 4; ++j) {
    const v8f acc = (j == 0) ? acc0 : (j == 1) ? acc1 : (j == 2) ? acc2 : acc3;
    const int n = nb + j * 16 + idx;
    const float bv = bias[n];
#pragma unroll
    for (int r = 0; r < 8; ++r) {
      const int m = mt * 16 + r + 8 * half;   // C layout: M = r + 8*(lane/16)
      float v = acc[r] + bv;
      if (act) v = fmaxf(v, 0.0f);
      Out[(size_t)m * N + n] = f2bf(v);
    }
  }
}

// LSTM gate GEMM: G[64,N](f32) = A0@W0^T + A1@W1^T + b0 + b1.
// Fused input + recurrent matmul. grid.x = N/64; 4 waves cover M=64.
__global__ void gates_gemm_kernel(const unsigned short* __restrict__ A0, int K0,
                                  const unsigned short* __restrict__ A1, int K1,
                                  const unsigned short* __restrict__ W0,
                                  const unsigned short* __restrict__ W1,
                                  const float* __restrict__ b0,
                                  const float* __restrict__ b1,
                                  float* __restrict__ Gout, int N) {
  const int lane = threadIdx.x & 31;
  const int half = lane >> 4, idx = lane & 15;
  const int mt   = threadIdx.x >> 5;       // 0..3 -> M=64
  const int nb   = blockIdx.x * 64;
  const int ko   = half * 8;

  v8f acc0 = {}, acc1 = {}, acc2 = {}, acc3 = {};
  {                                         // segment 0: x @ Wih^T
    const u32x4* pa  = (const u32x4*)(A0 + (size_t)(mt * 16 + idx) * K0 + ko);
    const u32x4* pb0 = (const u32x4*)(W0 + (size_t)(nb + 0  + idx) * K0 + ko);
    const u32x4* pb1 = (const u32x4*)(W0 + (size_t)(nb + 16 + idx) * K0 + ko);
    const u32x4* pb2 = (const u32x4*)(W0 + (size_t)(nb + 32 + idx) * K0 + ko);
    const u32x4* pb3 = (const u32x4*)(W0 + (size_t)(nb + 48 + idx) * K0 + ko);
    gemm_strip(pa, pb0, pb1, pb2, pb3, K0, acc0, acc1, acc2, acc3);
  }
  {                                         // segment 1: h @ Whh^T
    const u32x4* pa  = (const u32x4*)(A1 + (size_t)(mt * 16 + idx) * K1 + ko);
    const u32x4* pb0 = (const u32x4*)(W1 + (size_t)(nb + 0  + idx) * K1 + ko);
    const u32x4* pb1 = (const u32x4*)(W1 + (size_t)(nb + 16 + idx) * K1 + ko);
    const u32x4* pb2 = (const u32x4*)(W1 + (size_t)(nb + 32 + idx) * K1 + ko);
    const u32x4* pb3 = (const u32x4*)(W1 + (size_t)(nb + 48 + idx) * K1 + ko);
    gemm_strip(pa, pb0, pb1, pb2, pb3, K1, acc0, acc1, acc2, acc3);
  }
#pragma unroll
  for (int j = 0; j < 4; ++j) {
    const v8f acc = (j == 0) ? acc0 : (j == 1) ? acc1 : (j == 2) ? acc2 : acc3;
    const int n = nb + j * 16 + idx;
    const float bias = b0[n] + b1[n];
#pragma unroll
    for (int r = 0; r < 8; ++r) {
      const int m = mt * 16 + r + 8 * half;
      Gout[(size_t)m * N + n] = acc[r] + bias;
    }
  }
}

// Logits GEMM: Out[64, slice] = cdn[64,512] @ emb[N,512]^T + bcls.
// N=5000 is ragged: clamp B rows for loads (valid memory), mask stores.
__global__ void logits_gemm_kernel(const unsigned short* __restrict__ A,
                                   const unsigned short* __restrict__ Wemb,
                                   const float* __restrict__ bias,
                                   float* __restrict__ Out, size_t ldo,
                                   int N, int K) {
  const int lane = threadIdx.x & 31;
  const int half = lane >> 4, idx = lane & 15;
  const int mt   = threadIdx.x >> 5;       // 0..3 -> M=64
  const int nb   = blockIdx.x * 64;
  const int ko   = half * 8;

  const u32x4* pa = (const u32x4*)(A + (size_t)(mt * 16 + idx) * K + ko);
  int r0 = nb + 0  + idx; r0 = (r0 < N) ? r0 : (N - 1);
  int r1 = nb + 16 + idx; r1 = (r1 < N) ? r1 : (N - 1);
  int r2 = nb + 32 + idx; r2 = (r2 < N) ? r2 : (N - 1);
  int r3 = nb + 48 + idx; r3 = (r3 < N) ? r3 : (N - 1);
  const u32x4* pb0 = (const u32x4*)(Wemb + (size_t)r0 * K + ko);
  const u32x4* pb1 = (const u32x4*)(Wemb + (size_t)r1 * K + ko);
  const u32x4* pb2 = (const u32x4*)(Wemb + (size_t)r2 * K + ko);
  const u32x4* pb3 = (const u32x4*)(Wemb + (size_t)r3 * K + ko);

  v8f acc0 = {}, acc1 = {}, acc2 = {}, acc3 = {};
  gemm_strip(pa, pb0, pb1, pb2, pb3, K, acc0, acc1, acc2, acc3);

#pragma unroll
  for (int j = 0; j < 4; ++j) {
    const v8f acc = (j == 0) ? acc0 : (j == 1) ? acc1 : (j == 2) ? acc2 : acc3;
    const int n = nb + j * 16 + idx;
    if (n < N) {
      const float bv = bias[n];
#pragma unroll
      for (int r = 0; r < 8; ++r) {
        const int m = mt * 16 + r + 8 * half;
        Out[(size_t)m * ldo + n] = acc[r] + bv;
      }
    }
  }
}

// LSTM pointwise: gates [B,4H] (i,f,g,o) -> h,c (fp32) + bf16 copies
__global__ void lstm_pw_kernel(const float* __restrict__ Gin,
                               float* __restrict__ h, float* __restrict__ c,
                               unsigned short* __restrict__ hb,
                               unsigned short* __restrict__ hcat) {
  const int i = blockIdx.x * blockDim.x + threadIdx.x;   // 0..B*H-1
  const int b = i >> 9, j = i & 511;
  const float* g = Gin + (size_t)b * G4_;
  const float ig = sigmoidf_(g[j]);
  const float fg = sigmoidf_(g[512 + j]);
  const float gg = tanhf(g[1024 + j]);
  const float og = sigmoidf_(g[1536 + j]);
  const float cn = fg * c[i] + ig * gg;
  const float hn = og * tanhf(cn);
  c[i] = cn;
  h[i] = hn;
  const unsigned short hv = f2bf(hn);
  hb[i] = hv;
  if (hcat) hcat[(size_t)b * 1024 + j] = hv;   // [h2 | ctx] concat buffer
}

// Build xcat[b] = [ emb[token] | ctx[b] ] in bf16 (teacher forcing, SOS at t=0)
__global__ void prep_kernel(const int* __restrict__ y,
                            const unsigned short* __restrict__ emb,
                            const float* __restrict__ ctx,
                            unsigned short* __restrict__ xcat, int t) {
  const int i = blockIdx.x * blockDim.x + threadIdx.x;   // 0..B*H-1
  const int b = i >> 9, j = i & 511;
  const int tok = (t == 0) ? 0 : y[b * T_ + (t - 1)];
  xcat[(size_t)b * 1024 + j]       = emb[(size_t)tok * H_ + j];
  xcat[(size_t)b * 1024 + 512 + j] = f2bf(ctx[(size_t)b * H_ + j]);
}

// One block per batch element: q = h2@Wq^T+bq, softmax(q.K/sqrt(256)), ctx=attn.V
__global__ void attn_kernel(const float* __restrict__ h2,
                            const float* __restrict__ Wq,
                            const float* __restrict__ bq,
                            const unsigned short* __restrict__ Kb,
                            const unsigned short* __restrict__ Vb,
                            float* __restrict__ ctx,
                            unsigned short* __restrict__ hcat) {
  __shared__ float h2s[H_];
  __shared__ float qs[HK_];
  __shared__ float sc[S_];
  __shared__ float red[256];
  const int b = blockIdx.x;
  const int tid = threadIdx.x;            // blockDim = 256

  for (int i = tid; i < H_; i += 256) h2s[i] = h2[(size_t)b * H_ + i];
  __syncthreads();

  {                                        // q projection: thread -> one of 256 outs
    const float* w = Wq + (size_t)tid * H_;
    float acc = bq[tid];
    for (int k = 0; k < H_; ++k) acc += h2s[k] * w[k];
    qs[tid] = acc;
  }
  __syncthreads();

  // scores: 8 bf16 per 16B load from L2-resident K
  const unsigned short* kb = Kb + (size_t)b * S_ * HK_;
  for (int i = tid; i < S_; i += 256) {
    const u32x4* kr = (const u32x4*)(kb + (size_t)i * HK_);
    float acc = 0.0f;
    for (int c = 0; c < HK_ / 8; ++c) {
      const u32x4 w = kr[c];
      const float* q8 = qs + c * 8;
#pragma unroll
      for (int e = 0; e < 4; ++e) {
        acc += q8[2 * e]     * bf2f((unsigned short)(w[e] & 0xFFFFu));
        acc += q8[2 * e + 1] * bf2f((unsigned short)(w[e] >> 16));
      }
    }
    sc[i] = acc * 0.0625f;                 // 1/sqrt(256)
  }
  __syncthreads();

  float lm = -3.4e38f;
  for (int i = tid; i < S_; i += 256) lm = fmaxf(lm, sc[i]);
  red[tid] = lm; __syncthreads();
  for (int s = 128; s > 0; s >>= 1) {
    if (tid < s) red[tid] = fmaxf(red[tid], red[tid + s]);
    __syncthreads();
  }
  const float mx = red[0];
  __syncthreads();

  float ls = 0.0f;
  for (int i = tid; i < S_; i += 256) {
    const float e = __expf(sc[i] - mx);
    sc[i] = e; ls += e;
  }
  red[tid] = ls; __syncthreads();
  for (int s = 128; s > 0; s >>= 1) {
    if (tid < s) red[tid] += red[tid + s];
    __syncthreads();
  }
  const float inv = 1.0f / red[0];
  __syncthreads();

  // context: each thread owns 2 adjacent channels, packed 2xbf16 loads from V
  const unsigned short* vb = Vb + (size_t)b * S_ * HV_;
  const int hh = tid * 2;                  // 256 threads cover HV=512
  float a0 = 0.0f, a1 = 0.0f;
  for (int i = 0; i < S_; ++i) {
    const unsigned int pv = *(const unsigned int*)(vb + (size_t)i * HV_ + hh);
    const float p = sc[i];
    a0 += p * bf2f((unsigned short)(pv & 0xFFFFu));
    a1 += p * bf2f((unsigned short)(pv >> 16));
  }
  a0 *= inv; a1 *= inv;                    // fold softmax normalization
  ctx[(size_t)b * HV_ + hh]     = a0;
  ctx[(size_t)b * HV_ + hh + 1] = a1;
  hcat[(size_t)b * 1024 + 512 + hh]     = f2bf(a0);
  hcat[(size_t)b * 1024 + 512 + hh + 1] = f2bf(a1);
}

// ---------------- Host launch ----------------
extern "C" void kernel_launch(void* const* d_in, const int* in_sizes, int n_in,
                              void* d_out, int out_size, void* d_ws, size_t ws_size,
                              hipStream_t stream) {
  const float* enc  = (const float*)d_in[0];
  const int*   y    = (const int*)  d_in[1];
  const float* embW = (const float*)d_in[2];
  const float* Wq   = (const float*)d_in[3];
  const float* bq   = (const float*)d_in[4];
  const float* Wk   = (const float*)d_in[5];
  const float* bk   = (const float*)d_in[6];
  const float* Wv   = (const float*)d_in[7];
  const float* bv   = (const float*)d_in[8];
  const float* Wih1 = (const float*)d_in[9];
  const float* bih1 = (const float*)d_in[10];
  const float* Whh1 = (const float*)d_in[11];
  const float* bhh1 = (const float*)d_in[12];
  const float* Wih2 = (const float*)d_in[13];
  const float* bih2 = (const float*)d_in[14];
  const float* Whh2 = (const float*)d_in[15];
  const float* bhh2 = (const float*)d_in[16];
  const float* Wcdn = (const float*)d_in[17];
  const float* bcdn = (const float*)d_in[18];
  const float* bcls = (const float*)d_in[19];
  float* out = (float*)d_out;

  char* base = (char*)d_ws;
  size_t off = 0;
  auto alloc = [&](size_t bytes) -> void* {
    void* p = base + off;
    off = (off + bytes + 255) & ~(size_t)255;
    return p;
  };

  // bf16 buffers (K+V ~100MB -> resident in 192MB L2 across all 300 steps)
  unsigned short* encb  = (unsigned short*)alloc((size_t)B_ * S_ * E_ * 2);
  unsigned short* Kb    = (unsigned short*)alloc((size_t)B_ * S_ * HK_ * 2);
  unsigned short* Vb    = (unsigned short*)alloc((size_t)B_ * S_ * HV_ * 2);
  unsigned short* embb  = (unsigned short*)alloc((size_t)V_ * H_ * 2);
  unsigned short* Wkb   = (unsigned short*)alloc((size_t)HK_ * E_ * 2);
  unsigned short* Wvb   = (unsigned short*)alloc((size_t)HV_ * E_ * 2);
  unsigned short* Wih1b = (unsigned short*)alloc((size_t)G4_ * 1024 * 2);
  unsigned short* Whh1b = (unsigned short*)alloc((size_t)G4_ * H_ * 2);
  unsigned short* Wih2b = (unsigned short*)alloc((size_t)G4_ * H_ * 2);
  unsigned short* Whh2b = (unsigned short*)alloc((size_t)G4_ * H_ * 2);
  unsigned short* Wcdnb = (unsigned short*)alloc((size_t)H_ * 1024 * 2);
  unsigned short* xcat  = (unsigned short*)alloc((size_t)B_ * 1024 * 2);
  unsigned short* h1b   = (unsigned short*)alloc((size_t)B_ * H_ * 2);
  unsigned short* h2b   = (unsigned short*)alloc((size_t)B_ * H_ * 2);
  unsigned short* hc    = (unsigned short*)alloc((size_t)B_ * 1024 * 2);
  unsigned short* cdnb  = (unsigned short*)alloc((size_t)B_ * H_ * 2);
  float* h1  = (float*)alloc((size_t)B_ * H_ * 4);
  float* c1  = (float*)alloc((size_t)B_ * H_ * 4);
  float* h2  = (float*)alloc((size_t)B_ * H_ * 4);
  float* c2  = (float*)alloc((size_t)B_ * H_ * 4);
  float* ctx = (float*)alloc((size_t)B_ * H_ * 4);
  float* G   = (float*)alloc((size_t)B_ * G4_ * 4);

  // ---- one-time casts to bf16 ----
  cast_f32_bf16_kernel<<<4096, 256, 0, stream>>>(enc,  encb,  (size_t)B_ * S_ * E_);
  cast_f32_bf16_kernel<<<2048, 256, 0, stream>>>(embW, embb,  (size_t)V_ * H_);
  cast_f32_bf16_kernel<<<512,  256, 0, stream>>>(Wk,   Wkb,   (size_t)HK_ * E_);
  cast_f32_bf16_kernel<<<512,  256, 0, stream>>>(Wv,   Wvb,   (size_t)HV_ * E_);
  cast_f32_bf16_kernel<<<1024, 256, 0, stream>>>(Wih1, Wih1b, (size_t)G4_ * 1024);
  cast_f32_bf16_kernel<<<1024, 256, 0, stream>>>(Whh1, Whh1b, (size_t)G4_ * H_);
  cast_f32_bf16_kernel<<<1024, 256, 0, stream>>>(Wih2, Wih2b, (size_t)G4_ * H_);
  cast_f32_bf16_kernel<<<1024, 256, 0, stream>>>(Whh2, Whh2b, (size_t)G4_ * H_);
  cast_f32_bf16_kernel<<<512,  256, 0, stream>>>(Wcdn, Wcdnb, (size_t)H_ * 1024);

  // ---- zero recurrent state (every call: ws is not re-poisoned between replays) ----
  hipMemsetAsync(h1,  0, (size_t)B_ * H_ * 4, stream);
  hipMemsetAsync(c1,  0, (size_t)B_ * H_ * 4, stream);
  hipMemsetAsync(h2,  0, (size_t)B_ * H_ * 4, stream);
  hipMemsetAsync(c2,  0, (size_t)B_ * H_ * 4, stream);
  hipMemsetAsync(ctx, 0, (size_t)B_ * H_ * 4, stream);
  hipMemsetAsync(h1b, 0, (size_t)B_ * H_ * 2, stream);
  hipMemsetAsync(h2b, 0, (size_t)B_ * H_ * 2, stream);
  hipMemsetAsync(hc,  0, (size_t)B_ * 1024 * 2, stream);

  // ---- KV projection (one-time big WMMA GEMMs, M = B*S = 65536) ----
  gemm_bf16out_kernel<<<dim3(HK_ / 64, (B_ * S_) / 64), 128, 0, stream>>>(
      encb, Wkb, bk, Kb, B_ * S_, HK_, E_, 0);
  gemm_bf16out_kernel<<<dim3(HV_ / 64, (B_ * S_) / 64), 128, 0, stream>>>(
      encb, Wvb, bv, Vb, B_ * S_, HV_, E_, 0);

  // ---- sequential decode, T=300 steps ----
  for (int t = 0; t < T_; ++t) {
    prep_kernel<<<128, 256, 0, stream>>>(y, embb, ctx, xcat, t);
    gates_gemm_kernel<<<G4_ / 64, 128, 0, stream>>>(
        xcat, 1024, h1b, H_, Wih1b, Whh1b, bih1, bhh1, G, G4_);
    lstm_pw_kernel<<<128, 256, 0, stream>>>(G, h1, c1, h1b, (unsigned short*)nullptr);
    gates_gemm_kernel<<<G4_ / 64, 128, 0, stream>>>(
        h1b, H_, h2b, H_, Wih2b, Whh2b, bih2, bhh2, G, G4_);
    lstm_pw_kernel<<<128, 256, 0, stream>>>(G, h2, c2, h2b, hc);
    attn_kernel<<<B_, 256, 0, stream>>>(h2, Wq, bq, Kb, Vb, ctx, hc);
    gemm_bf16out_kernel<<<dim3(H_ / 64, 1), 128, 0, stream>>>(
        hc, Wcdnb, bcdn, cdnb, B_, H_, 1024, 1 /*relu*/);
    logits_gemm_kernel<<<dim3((V_ + 63) / 64, 1), 128, 0, stream>>>(
        cdnb, embb, bcls, out + (size_t)t * V_, (size_t)T_ * V_, V_, H_);
  }
}
